// MessagePassingConv_14078902796825
// MI455X (gfx1250) — matrix operand order repaired and checked by reading, not compiled, not run on hardware
//
#include <hip/hip_runtime.h>
#include <math.h>

#define NNODES 500000
#define NEDGES 500000
#define FDIM   128
#define BN_EPS 1e-3f

typedef __attribute__((ext_vector_type(16))) __bf16 v16bf;
typedef __attribute__((ext_vector_type(8)))  float  v8f;

// ---------------------------------------------------------------------------
// Helpers: build a 16x32 bf16 A-fragment from an f32 row in memory.
// Per ISA 7.12.2 (16-bit A-matrix 16x32): lane L holds row M = L&15.
// Lanes 0-15:  elems 0..7 -> K=kbase+0..7,   elems 8..15 -> K=kbase+16..23
// Lanes 16-31: elems 0..7 -> K=kbase+8..15,  elems 8..15 -> K=kbase+24..31
// i.e. two contiguous runs of 8 floats at kbase+s and kbase+16+s, s=8*(lane>>4).
// ---------------------------------------------------------------------------
__device__ __forceinline__ v16bf load_a_frag(const float* __restrict__ row,
                                             int kbase, int lane) {
  const int s = (lane >> 4) * 8;
  const float* p0 = row + kbase + s;
  const float* p1 = row + kbase + 16 + s;
  v16bf r;
#pragma unroll
  for (int i = 0; i < 8; ++i) r[i] = (__bf16)p0[i];
#pragma unroll
  for (int i = 0; i < 8; ++i) r[8 + i] = (__bf16)p1[i];
  return r;
}

__device__ __forceinline__ v16bf load_a_frag_scaled(const float* __restrict__ row,
                                                    int kbase, int lane,
                                                    const float* __restrict__ sc,
                                                    const float* __restrict__ sh) {
  const int s = (lane >> 4) * 8;
  const int k0 = kbase + s, k1 = kbase + 16 + s;
  v16bf r;
#pragma unroll
  for (int i = 0; i < 8; ++i) r[i] = (__bf16)fmaf(row[k0 + i], sc[k0 + i], sh[k0 + i]);
#pragma unroll
  for (int i = 0; i < 8; ++i) r[8 + i] = (__bf16)fmaf(row[k1 + i], sc[k1 + i], sh[k1 + i]);
  return r;
}

__device__ __forceinline__ float sigmoidf_(float v) {
  return 1.0f / (1.0f + __expf(-v));
}

// ---------------------------------------------------------------------------
// K0: repack an f32 weight W[K=128][ncols] into bf16 WMMA B-fragments.
// B-matrix 32x16 bf16 layout (matches A): lane L holds column N = L&15;
// lanes 0-15 hold K=ktile*32+{0..7,16..23}, lanes 16-31 hold K=ktile*32+{8..15,24..31}.
// Fragment storage: frag[((ntile*4 + ktile)*32 + lane)*16 + j]
// ---------------------------------------------------------------------------
__global__ void repack_b_kernel(const float* __restrict__ W,
                                __bf16* __restrict__ frag, int ncols) {
  int idx = blockIdx.x * blockDim.x + threadIdx.x;
  int total = (ncols >> 4) * 4 * 32 * 16;
  if (idx >= total) return;
  int j     = idx & 15;
  int lane  = (idx >> 4) & 31;
  int ktile = (idx >> 9) & 3;
  int ntile = idx >> 11;
  int k = ktile * 32 + ((j >> 3) * 16) + ((lane >> 4) * 8) + (j & 7);
  int n = ntile * 16 + (lane & 15);
  frag[idx] = (__bf16)W[k * ncols + n];
}

// ---------------------------------------------------------------------------
// K1: edge scatter-add. One wave32 per edge; each lane moves float4 (4 cols).
// pairs[e] = {dst, src}; accum[dst] += x[src]
// ---------------------------------------------------------------------------
__global__ __launch_bounds__(256) void scatter_kernel(const float* __restrict__ x,
                                                      const int* __restrict__ pairs,
                                                      float* __restrict__ accum,
                                                      int nEdges) {
  int wave = blockIdx.x * (blockDim.x >> 5) + (threadIdx.x >> 5);
  int lane = threadIdx.x & 31;
  if (wave >= nEdges) return;
  int dst = pairs[wave * 2 + 0];
  int src = pairs[wave * 2 + 1];
  const float4 v = ((const float4*)(x + (long)src * FDIM))[lane];
  float* d = accum + (long)dst * FDIM + lane * 4;
  atomicAdd(d + 0, v.x);
  atomicAdd(d + 1, v.y);
  atomicAdd(d + 2, v.z);
  atomicAdd(d + 3, v.w);
}

// ---------------------------------------------------------------------------
// K2: aggre = sum_next @ Wn + sum_prev @ Wp + b + x ; relu ; write a -> out
//     + per-column sum / sumsq (LDS reduce, then global atomics).
// 256 threads = 8 waves; each wave owns 16 rows; 8 column tiles of 16.
// ---------------------------------------------------------------------------
__global__ __launch_bounds__(256) void aggre_kernel(
    const float* __restrict__ x,
    const float* __restrict__ sum_next,
    const float* __restrict__ sum_prev,
    const v16bf* __restrict__ wnF,
    const v16bf* __restrict__ wpF,
    const float* __restrict__ bias,
    float* __restrict__ a_out,
    float* __restrict__ colsum,
    float* __restrict__ colsq) {
  __shared__ float s_sum[FDIM];
  __shared__ float s_sq[FDIM];
  const int tid = threadIdx.x;
  if (tid < FDIM) { s_sum[tid] = 0.0f; s_sq[tid] = 0.0f; }
  __syncthreads();

  const int wave = tid >> 5;
  const int lane = tid & 31;
  const long rowBase = (long)blockIdx.x * 128 + wave * 16;
  if (rowBase < NNODES) {
    const int m = lane & 15;
    const float* rn = sum_next + (rowBase + m) * FDIM;
    const float* rp = sum_prev + (rowBase + m) * FDIM;
    v16bf an[4], ap[4];
#pragma unroll
    for (int kc = 0; kc < 4; ++kc) {
      an[kc] = load_a_frag(rn, kc * 32, lane);
      ap[kc] = load_a_frag(rp, kc * 32, lane);
    }
    const int col0 = lane & 15;
    const int rOff = (lane >> 4) * 8;
#pragma unroll
    for (int nt = 0; nt < 8; ++nt) {
      v8f c = {};
#pragma unroll
      for (int kc = 0; kc < 4; ++kc) {
        v16bf b1 = wnF[(nt * 4 + kc) * 32 + lane];
        c = __builtin_amdgcn_wmma_f32_16x16x32_bf16(false, an[kc], false, b1,
                                                    (short)0, c, false, false);
        v16bf b2 = wpF[(nt * 4 + kc) * 32 + lane];
        c = __builtin_amdgcn_wmma_f32_16x16x32_bf16(false, ap[kc], false, b2,
                                                    (short)0, c, false, false);
      }
      const int col = nt * 16 + col0;
      const float bb = bias[col];
      float ps = 0.0f, pq = 0.0f;
#pragma unroll
      for (int r = 0; r < 8; ++r) {
        const long row = rowBase + rOff + r;
        float v = c[r] + bb + x[row * FDIM + col];
        v = v > 0.0f ? v : 0.0f;              // relu
        a_out[row * FDIM + col] = v;
        ps += v;
        pq += v * v;
      }
      atomicAdd(&s_sum[col], ps);
      atomicAdd(&s_sq[col], pq);
    }
  }
  __syncthreads();
  if (tid < FDIM) {
    atomicAdd(&colsum[tid], s_sum[tid]);
    atomicAdd(&colsq[tid], s_sq[tid]);
  }
}

// ---------------------------------------------------------------------------
// K3: fold BN into per-column scale/shift: a_norm = a*scale + shift
// ---------------------------------------------------------------------------
__global__ void stats_kernel(const float* __restrict__ colsum,
                             const float* __restrict__ colsq,
                             const float* __restrict__ gamma,
                             const float* __restrict__ beta,
                             float* __restrict__ scaleArr,
                             float* __restrict__ shiftArr) {
  int c = threadIdx.x;
  if (c < FDIM) {
    const float invN = 1.0f / (float)NNODES;
    float mean = colsum[c] * invN;
    float var  = colsq[c] * invN - mean * mean;
    float inv  = rsqrtf(var + BN_EPS);
    float sc   = inv * gamma[c];
    scaleArr[c] = sc;
    shiftArr[c] = beta[c] - mean * sc;
  }
}

// ---------------------------------------------------------------------------
// K4: GRU. matrix_x = a_norm @ K(128x384)+b0 ; matrix_inner = x @ R +b1.
// Gate columns: z = [0,128), r = [128,256), h = [256,384).
// Per wave: 16 rows, all A frags resident; 8 feature tiles x 6 accumulators.
// Reads a (= out) for its own rows before overwriting them with the result.
// ---------------------------------------------------------------------------
__global__ __launch_bounds__(256) void gru_kernel(
    const float* __restrict__ x,
    const v16bf* __restrict__ gkF,
    const v16bf* __restrict__ grkF,
    const float* __restrict__ gbias,      // [2][384]
    const float* __restrict__ scaleArr,
    const float* __restrict__ shiftArr,
    float* out) {
  const int tid  = threadIdx.x;
  const int wave = tid >> 5;
  const int lane = tid & 31;
  const long rowBase = (long)blockIdx.x * 128 + wave * 16;
  if (rowBase >= NNODES) return;

  const int m = lane & 15;
  const float* arow = out + (rowBase + m) * FDIM;
  const float* xrow = x + (rowBase + m) * FDIM;
  v16bf aA[4], xA[4];
#pragma unroll
  for (int kc = 0; kc < 4; ++kc) {
    aA[kc] = load_a_frag_scaled(arow, kc * 32, lane, scaleArr, shiftArr);
    xA[kc] = load_a_frag(xrow, kc * 32, lane);
  }

  const int col0 = lane & 15;
  const int rOff = (lane >> 4) * 8;
#pragma unroll
  for (int ft = 0; ft < 8; ++ft) {
    v8f cz = {}, cr = {}, ch = {}, dz = {}, dr = {}, dh = {};
#pragma unroll
    for (int kc = 0; kc < 4; ++kc) {
      v16bf bz = gkF[((ft) * 4 + kc) * 32 + lane];
      cz = __builtin_amdgcn_wmma_f32_16x16x32_bf16(false, aA[kc], false, bz,
                                                   (short)0, cz, false, false);
      v16bf br = gkF[((ft + 8) * 4 + kc) * 32 + lane];
      cr = __builtin_amdgcn_wmma_f32_16x16x32_bf16(false, aA[kc], false, br,
                                                   (short)0, cr, false, false);
      v16bf bh = gkF[((ft + 16) * 4 + kc) * 32 + lane];
      ch = __builtin_amdgcn_wmma_f32_16x16x32_bf16(false, aA[kc], false, bh,
                                                   (short)0, ch, false, false);
      v16bf ez = grkF[((ft) * 4 + kc) * 32 + lane];
      dz = __builtin_amdgcn_wmma_f32_16x16x32_bf16(false, xA[kc], false, ez,
                                                   (short)0, dz, false, false);
      v16bf er = grkF[((ft + 8) * 4 + kc) * 32 + lane];
      dr = __builtin_amdgcn_wmma_f32_16x16x32_bf16(false, xA[kc], false, er,
                                                   (short)0, dr, false, false);
      v16bf eh = grkF[((ft + 16) * 4 + kc) * 32 + lane];
      dh = __builtin_amdgcn_wmma_f32_16x16x32_bf16(false, xA[kc], false, eh,
                                                   (short)0, dh, false, false);
    }
    const int f = ft * 16 + col0;  // feature column 0..127
    const float b0z = gbias[f],        b0r = gbias[128 + f],       b0h = gbias[256 + f];
    const float b1z = gbias[384 + f],  b1r = gbias[384 + 128 + f], b1h = gbias[384 + 256 + f];
#pragma unroll
    for (int r = 0; r < 8; ++r) {
      const long row = rowBase + rOff + r;
      const float h  = x[row * FDIM + f];
      const float z  = sigmoidf_((cz[r] + b0z) + (dz[r] + b1z));
      const float rg = sigmoidf_((cr[r] + b0r) + (dr[r] + b1r));
      const float hc = tanhf((ch[r] + b0h) + rg * (dh[r] + b1h));
      out[row * FDIM + f] = z * h + (1.0f - z) * hc;
    }
  }
}

// ---------------------------------------------------------------------------
extern "C" void kernel_launch(void* const* d_in, const int* in_sizes, int n_in,
                              void* d_out, int out_size, void* d_ws, size_t ws_size,
                              hipStream_t stream) {
  const float* x          = (const float*)d_in[0];
  const int*   pairs_prev = (const int*)d_in[1];
  const int*   pairs_next = (const int*)d_in[2];
  const float* w_next     = (const float*)d_in[3];
  const float* w_prev     = (const float*)d_in[4];
  const float* b          = (const float*)d_in[5];
  const float* bn_gamma   = (const float*)d_in[6];
  const float* bn_beta    = (const float*)d_in[7];
  const float* gru_k      = (const float*)d_in[8];
  const float* gru_rk     = (const float*)d_in[9];
  const float* gru_b      = (const float*)d_in[10];
  float* out = (float*)d_out;

  const size_t nf = (size_t)NNODES * FDIM;
  char* ws = (char*)d_ws;
  float* sum_next = (float*)ws; ws += nf * sizeof(float);
  float* sum_prev = (float*)ws; ws += nf * sizeof(float);
  float* colsum   = (float*)ws; ws += FDIM * sizeof(float);
  float* colsq    = (float*)ws; ws += FDIM * sizeof(float);
  float* scaleArr = (float*)ws; ws += FDIM * sizeof(float);
  float* shiftArr = (float*)ws; ws += FDIM * sizeof(float);
  __bf16* wnF  = (__bf16*)ws; ws += (size_t)128 * 128 * 2;
  __bf16* wpF  = (__bf16*)ws; ws += (size_t)128 * 128 * 2;
  __bf16* gkF  = (__bf16*)ws; ws += (size_t)128 * 384 * 2;
  __bf16* grkF = (__bf16*)ws; ws += (size_t)128 * 384 * 2;

  hipMemsetAsync(sum_next, 0, nf * sizeof(float), stream);
  hipMemsetAsync(sum_prev, 0, nf * sizeof(float), stream);
  hipMemsetAsync(colsum, 0, 2 * FDIM * sizeof(float), stream);  // colsum+colsq

  repack_b_kernel<<<(128 * 128 + 255) / 256, 256, 0, stream>>>(w_next, wnF, 128);
  repack_b_kernel<<<(128 * 128 + 255) / 256, 256, 0, stream>>>(w_prev, wpF, 128);
  repack_b_kernel<<<(128 * 384 + 255) / 256, 256, 0, stream>>>(gru_k, gkF, 384);
  repack_b_kernel<<<(128 * 384 + 255) / 256, 256, 0, stream>>>(gru_rk, grkF, 384);

  scatter_kernel<<<(NEDGES + 7) / 8, 256, 0, stream>>>(x, pairs_next, sum_next, NEDGES);
  scatter_kernel<<<(NEDGES + 7) / 8, 256, 0, stream>>>(x, pairs_prev, sum_prev, NEDGES);

  const int rowBlocks = (NNODES + 127) / 128;
  aggre_kernel<<<rowBlocks, 256, 0, stream>>>(x, sum_next, sum_prev,
                                              (const v16bf*)wnF, (const v16bf*)wpF,
                                              b, out, colsum, colsq);
  stats_kernel<<<1, 128, 0, stream>>>(colsum, colsq, bn_gamma, bn_beta,
                                      scaleArr, shiftArr);
  gru_kernel<<<rowBlocks, 256, 0, stream>>>(x, (const v16bf*)gkF, (const v16bf*)grkF,
                                            gru_b, scaleArr, shiftArr, out);
}
